// Decoder_72258529788345
// MI455X (gfx1250) — matrix operand order repaired and checked by reading, not compiled, not run on hardware
//
#include <hip/hip_runtime.h>
#include <hip/hip_bf16.h>
#include <math.h>

// ---------------------------------------------------------------------------
// 2-layer LSTM decoder for MI455X (gfx1250).
//   - bf16 WMMA (v_wmma_f32_16x16x32_bf16) for all GEMMs, fp32 accumulate.
//   - Weights pre-transposed to K-major bf16 in d_ws (fits easily in 192MB L2).
//   - B-operand slabs staged global->LDS with CDNA5 async copy
//     (global_load_async_to_lds_b128, ASYNCcnt) instead of VGPR round-trips.
//   - Grid: 8 batch tiles (64 rows) x 16 H-column groups (64 units) = 128 WGs,
//     256 threads (8 wave32) each. Recurrence handled by per-batch-tile
//     global-atomic split barriers (2 per timestep).
// ---------------------------------------------------------------------------

#define B_    512
#define T_    168
#define F_    32
#define H_    1024
#define GE_   16
#define IN0_  49
#define K0PAD 64
#define G4H   4096

#define NBT     8     // batch tiles
#define BTILE   64
#define CG      16    // column groups per tile
#define NTHREADS 256
#define NBAR   (T_ * 2 * NBT)

typedef __attribute__((ext_vector_type(16))) __bf16 v16bf;
typedef __attribute__((ext_vector_type(8)))  float  v8f;

union Frag16 { v16bf v; uint4 q[2]; };

static __device__ __forceinline__ unsigned short f2bf(float f) {
  unsigned int u = __float_as_uint(f);
  unsigned int r = (u + 0x7FFFu + ((u >> 16) & 1u)) >> 16;
  return (unsigned short)r;
}
static __device__ __forceinline__ float sigf(float x) {
  return 1.0f / (1.0f + __expf(-x));
}
static __device__ __forceinline__ float tanh_fast(float x) {
  return 1.0f - 2.0f / (__expf(2.0f * x) + 1.0f);
}

struct Params {
  // ws-resident
  const unsigned short* wt_ih0;   // [K0PAD][4096] k-major bf16
  const unsigned short* wt_hh0;   // [1024][4096]
  const unsigned short* wt_ih1;   // [1024][4096]
  const unsigned short* wt_hh1;   // [1024][4096]
  unsigned short* h0buf[2];       // ping-pong [B][H] bf16
  unsigned short* h1buf[2];
  float* c0;                      // [B][H] fp32 (in-place per-slice update)
  float* c1;
  const float* gv;                // [B][GE]
  const float* bias0;             // [4096] = b_ih0 + b_hh0
  const float* bias1;
  float* predpart;                // [B][CG]
  unsigned int* bar;              // [T][2][NBT]
  // inputs
  const float* dec_known;         // [B][T][F]
  const float* target_y;          // [B][T]
  const float* last_enc;          // [B]
  const float* W_proj;            // [H]
  const float* b_proj;            // [1]
  const int*   tf;                // [T]
  // output
  float* out;                     // [B][T]
};

static __device__ __forceinline__ void tile_barrier(unsigned int* slot) {
  __threadfence();
  __syncthreads();
  if (threadIdx.x == 0) {
    atomicAdd(slot, 1u);
    while (atomicAdd(slot, 0u) < (unsigned)CG) __builtin_amdgcn_s_sleep(1);
  }
  __syncthreads();
  __threadfence();
}

// Accumulate A(64 x K) * WTslice(K x 256) into per-wave acc[8] tiles.
// WT is k-major [Krows][4096]; this WG's 256 output cols map to the 4 gate
// blocks: gcol(c) = (c/64)*1024 + ncolBase + (c%64).
static __device__ __forceinline__ void gemm_bf16(
    v8f acc[8],
    const unsigned short* __restrict__ WT, int ncolBase,
    const unsigned short* __restrict__ A, int lda, int K,
    unsigned short* sB, int tid) {
  const int lane = tid & 31;
  const int wave = tid >> 5;
  const int mt = wave >> 1;            // M tile 0..3
  const int ntBase = (wave & 1) * 8;   // N tiles [ntBase, ntBase+8)
  const int arow = lane & 15;
  const int asel = (lane >> 4) << 3;   // K sub-chunk select per A layout
  // Flat address low 32 bits of an LDS-aperture pointer == LDS byte offset
  // (ISA 10.2 aperture rules); usable as the async-copy LDS dest VGPR.
  const unsigned ldsBase = (unsigned)(size_t)sB;
  for (int kb = 0; kb < K; kb += 32) {
    // Async-stage B slab (32 x 256 bf16 = 16KB) into LDS: 16B per lane per
    // issue via GLOBAL_LOAD_ASYNC_TO_LDS_B128 (ASYNCcnt), no VGPR round-trip.
    // Constant trip count (4) so no exec-mask guards are generated.
#pragma unroll
    for (int it = 0; it < 4; ++it) {
      int i = tid + it * NTHREADS;       // 0..1023, 16B chunk index
      int e = i << 3;                    // bf16 element index, 8 per chunk
      int r = e >> 8;
      int c = e & 255;
      int gcol = ((c >> 6) << 10) + ncolBase + (c & 63);
      const unsigned short* gp = WT + (size_t)(kb + r) * G4H + gcol;
      unsigned ldsOff = ldsBase + (unsigned)(((r << 8) + c) << 1);
      asm volatile("global_load_async_to_lds_b128 %0, %1, off"
                   :: "v"(ldsOff), "v"(gp)
                   : "memory");
    }
    asm volatile("s_wait_asynccnt 0x0" ::: "memory");
    __syncthreads();
    // A fragment: lane L holds row M=L%16, halves 0..7 = K[kb+sel..+8),
    // halves 8..15 = K[kb+16+sel..+8)  (ISA 7.12.2, 16-bit A 16x32).
    Frag16 a;
    const unsigned short* ap = A + (size_t)((mt << 4) + arow) * lda + kb + asel;
    a.q[0] = *(const uint4*)(ap);
    a.q[1] = *(const uint4*)(ap + 16);
#pragma unroll
    for (int j = 0; j < 8; ++j) {
      // B fragment: lane L holds column N=L%16; halves 0..7 = K[kb+sel..+8),
      // halves 8..15 = K[kb+16+sel..+8) (same K order as A).
      Frag16 b;
      const unsigned short* bp = sB + (asel << 8) + ((ntBase + j) << 4) + arow;
#pragma unroll
      for (int s = 0; s < 16; ++s)
        b.v[s] = __builtin_bit_cast(__bf16, bp[(s < 8 ? s : s + 8) << 8]);
      acc[j] = __builtin_amdgcn_wmma_f32_16x16x32_bf16(
          false, a.v, false, b.v, (short)0, acc[j], false, false);
    }
    __syncthreads();
  }
}

// C/D layout: vgpr r, lane L -> (M = r + 8*(L/16), N = L%16).
static __device__ __forceinline__ void dump_acc(const v8f acc[8], float* sG,
                                                int tid) {
  const int lane = tid & 31;
  const int wave = tid >> 5;
  const int mt = wave >> 1;
  const int ntBase = (wave & 1) * 8;
  const int m0 = (mt << 4) + ((lane >> 4) << 3);
  const int n = lane & 15;
#pragma unroll
  for (int j = 0; j < 8; ++j) {
#pragma unroll
    for (int r = 0; r < 8; ++r)
      sG[(m0 + r) * 256 + ((ntBase + j) << 4) + n] = acc[j][r];
  }
}

// Dynamic LDS partition (bytes):
//   sB  : 32*256 bf16  = 16384
//   sX  : 64*64  bf16  =  8192
//   sG  : 64*256 f32   = 65536
//   sH1 : 64*64  f32   = 16384
//   sPY : 64     f32   =   256
#define SMEM_BYTES (16384 + 8192 + 65536 + 16384 + 256)

__global__ __launch_bounds__(NTHREADS) void lstm_main(Params p) {
  const int tid = threadIdx.x;
  const int tile = blockIdx.x & 7;
  const int grp = blockIdx.x >> 3;
  const int bbase = tile * BTILE;
  const int ncb = grp * 64;

  extern __shared__ char smem[];
  unsigned short* sB = (unsigned short*)smem;
  unsigned short* sX = (unsigned short*)(smem + 16384);
  float* sG = (float*)(smem + 16384 + 8192);
  float* sH1 = (float*)(smem + 16384 + 8192 + 65536);
  float* sPY = (float*)(smem + 16384 + 8192 + 65536 + 16384);

  if (tid < 64) sPY[tid] = p.last_enc[bbase + tid];
  __syncthreads();

  for (int t = 0; t < T_; ++t) {
    const unsigned short* h0cur = p.h0buf[t & 1];
    unsigned short* h0nxt = p.h0buf[(t + 1) & 1];
    const unsigned short* h1cur = p.h1buf[t & 1];
    unsigned short* h1nxt = p.h1buf[(t + 1) & 1];

    // Assemble layer-0 input x = [prev_y | known_t | gv | pad] as bf16.
#pragma unroll
    for (int it = 0; it < 16; ++it) {
      int idx = tid + it * NTHREADS;
      int bl = idx >> 6, k = idx & 63;
      float v = 0.0f;
      if (k == 0)
        v = sPY[bl];
      else if (k < 33)
        v = p.dec_known[((size_t)(bbase + bl) * T_ + t) * F_ + (k - 1)];
      else if (k < 49)
        v = p.gv[(bbase + bl) * GE_ + (k - 33)];
      sX[idx] = f2bf(v);
    }
    __syncthreads();

    // ---- layer 0 GEMM: gates0 = x @ W_ih0^T + h0 @ W_hh0^T ----
    v8f acc[8];
    v8f zero = {0.f, 0.f, 0.f, 0.f, 0.f, 0.f, 0.f, 0.f};
#pragma unroll
    for (int j = 0; j < 8; ++j) acc[j] = zero;
    gemm_bf16(acc, p.wt_ih0, ncb, sX, 64, K0PAD, sB, tid);
    gemm_bf16(acc, p.wt_hh0, ncb, h0cur + (size_t)bbase * H_, H_, H_, sB, tid);
    dump_acc(acc, sG, tid);
    __syncthreads();

    // ---- cell 0 ----
#pragma unroll 2
    for (int it = 0; it < 16; ++it) {
      int idx = tid + it * NTHREADS;
      int bl = idx >> 6, hl = idx & 63;
      int hg = ncb + hl;
      float ii = sG[bl * 256 + hl] + p.bias0[hg];
      float ff = sG[bl * 256 + 64 + hl] + p.bias0[1024 + hg];
      float gg = sG[bl * 256 + 128 + hl] + p.bias0[2048 + hg];
      float oo = sG[bl * 256 + 192 + hl] + p.bias0[3072 + hg];
      size_t ci = (size_t)(bbase + bl) * H_ + hg;
      float c = p.c0[ci];
      float cn = sigf(ff) * c + sigf(ii) * tanh_fast(gg);
      p.c0[ci] = cn;
      h0nxt[ci] = f2bf(sigf(oo) * tanh_fast(cn));
    }
    tile_barrier(p.bar + ((t * 2 + 0) * NBT + tile));  // h0n visible tile-wide

    // ---- layer 1 GEMM: gates1 = h0n @ W_ih1^T + h1 @ W_hh1^T ----
#pragma unroll
    for (int j = 0; j < 8; ++j) acc[j] = zero;
    gemm_bf16(acc, p.wt_ih1, ncb, h0nxt + (size_t)bbase * H_, H_, H_, sB, tid);
    gemm_bf16(acc, p.wt_hh1, ncb, h1cur + (size_t)bbase * H_, H_, H_, sB, tid);
    dump_acc(acc, sG, tid);
    __syncthreads();

    // ---- cell 1 (+ stash h1n fp32 for projection) ----
#pragma unroll 2
    for (int it = 0; it < 16; ++it) {
      int idx = tid + it * NTHREADS;
      int bl = idx >> 6, hl = idx & 63;
      int hg = ncb + hl;
      float ii = sG[bl * 256 + hl] + p.bias1[hg];
      float ff = sG[bl * 256 + 64 + hl] + p.bias1[1024 + hg];
      float gg = sG[bl * 256 + 128 + hl] + p.bias1[2048 + hg];
      float oo = sG[bl * 256 + 192 + hl] + p.bias1[3072 + hg];
      size_t ci = (size_t)(bbase + bl) * H_ + hg;
      float c = p.c1[ci];
      float cn = sigf(ff) * c + sigf(ii) * tanh_fast(gg);
      p.c1[ci] = cn;
      float hn = sigf(oo) * tanh_fast(cn);
      h1nxt[ci] = f2bf(hn);
      sH1[idx] = hn;
    }
    __syncthreads();

    // partial projection over this group's 64 hidden units
    if (tid < 64) {
      float s = 0.0f;
#pragma unroll 8
      for (int hl = 0; hl < 64; ++hl)
        s += sH1[tid * 64 + hl] * p.W_proj[ncb + hl];
      p.predpart[(bbase + tid) * CG + grp] = s;
    }
    tile_barrier(p.bar + ((t * 2 + 1) * NBT + tile));  // h1n + predpart done

    // reduce partials -> pred; teacher forcing -> prev_y for next step
    if (tid < 64) {
      float s = p.b_proj[0];
#pragma unroll
      for (int g2 = 0; g2 < CG; ++g2) s += p.predpart[(bbase + tid) * CG + g2];
      if (grp == 0) p.out[(size_t)(bbase + tid) * T_ + t] = s;
      sPY[tid] = p.tf[t] ? p.target_y[(size_t)(bbase + tid) * T_ + t] : s;
    }
    __syncthreads();
  }
}

// ------------------------------ prologue kernels ---------------------------

__global__ void k_zero32(unsigned int* p, int n) {
  int i = blockIdx.x * blockDim.x + threadIdx.x;
  if (i < n) p[i] = 0u;
}
__global__ void k_bias(const float* a, const float* b, float* o, int n) {
  int i = blockIdx.x * blockDim.x + threadIdx.x;
  if (i < n) o[i] = a[i] + b[i];
}
__global__ void k_gv(const float* emb, const int* gid, float* gv) {
  int i = blockIdx.x * blockDim.x + threadIdx.x;
  if (i < B_ * GE_) {
    int b = i >> 4, e = i & 15;
    gv[i] = emb[(size_t)gid[b] * GE_ + e];
  }
}
__global__ void k_state(const float* h0, const float* c0, unsigned short* h0A,
                        unsigned short* h1A, float* c0w, float* c1w) {
  int i = blockIdx.x * blockDim.x + threadIdx.x;
  if (i < B_ * H_) {
    h0A[i] = f2bf(h0[i]);
    h1A[i] = f2bf(h0[B_ * H_ + i]);
    c0w[i] = c0[i];
    c1w[i] = c0[B_ * H_ + i];
  }
}
// W (4096 x K) row-major -> WT (Kpad x 4096) k-major bf16, zero-padded K.
__global__ void k_wt(const float* W, unsigned short* out, int K, int n) {
  int i = blockIdx.x * blockDim.x + threadIdx.x;
  if (i < n) {
    int k = i >> 12;
    int col = i & 4095;
    out[i] = (k < K) ? f2bf(W[(size_t)col * K + k]) : (unsigned short)0;
  }
}

// ------------------------------ launcher -----------------------------------

extern "C" void kernel_launch(void* const* d_in, const int* in_sizes, int n_in,
                              void* d_out, int out_size, void* d_ws,
                              size_t ws_size, hipStream_t stream) {
  (void)in_sizes; (void)n_in; (void)out_size;
  const float* dec_known = (const float*)d_in[0];
  const float* target_y  = (const float*)d_in[1];
  const float* last_enc  = (const float*)d_in[2];
  const float* h0in      = (const float*)d_in[3];
  const float* c0in      = (const float*)d_in[4];
  const float* gemb      = (const float*)d_in[5];
  const float* W_ih0     = (const float*)d_in[6];
  const float* W_hh0     = (const float*)d_in[7];
  const float* b_ih0     = (const float*)d_in[8];
  const float* b_hh0     = (const float*)d_in[9];
  const float* W_ih1     = (const float*)d_in[10];
  const float* W_hh1     = (const float*)d_in[11];
  const float* b_ih1     = (const float*)d_in[12];
  const float* b_hh1     = (const float*)d_in[13];
  const float* W_proj    = (const float*)d_in[14];
  const float* b_proj    = (const float*)d_in[15];
  const int*   gids      = (const int*)d_in[16];
  const int*   tf        = (const int*)d_in[17];

  // ---- workspace layout (bytes) ----
  char* w = (char*)d_ws;
  size_t off = 0;
  auto take = [&](size_t bytes) {
    char* p = w + off;
    off = (off + bytes + 255) & ~(size_t)255;
    return p;
  };
  unsigned short* wt_ih0 = (unsigned short*)take((size_t)K0PAD * G4H * 2);
  unsigned short* wt_hh0 = (unsigned short*)take((size_t)H_ * G4H * 2);
  unsigned short* wt_ih1 = (unsigned short*)take((size_t)H_ * G4H * 2);
  unsigned short* wt_hh1 = (unsigned short*)take((size_t)H_ * G4H * 2);
  unsigned short* h0A = (unsigned short*)take((size_t)B_ * H_ * 2);
  unsigned short* h0B = (unsigned short*)take((size_t)B_ * H_ * 2);
  unsigned short* h1A = (unsigned short*)take((size_t)B_ * H_ * 2);
  unsigned short* h1B = (unsigned short*)take((size_t)B_ * H_ * 2);
  float* c0w = (float*)take((size_t)B_ * H_ * 4);
  float* c1w = (float*)take((size_t)B_ * H_ * 4);
  float* gv = (float*)take((size_t)B_ * GE_ * 4);
  float* bias0 = (float*)take((size_t)G4H * 4);
  float* bias1 = (float*)take((size_t)G4H * 4);
  float* predpart = (float*)take((size_t)B_ * CG * 4);
  unsigned int* bar = (unsigned int*)take((size_t)NBAR * 4);
  if (off > ws_size) return;  // workspace too small; nothing safe to do

  // ---- prologue: deterministic re-init every call ----
  k_zero32<<<(NBAR + 255) / 256, 256, 0, stream>>>(bar, NBAR);
  k_bias<<<G4H / 256, 256, 0, stream>>>(b_ih0, b_hh0, bias0, G4H);
  k_bias<<<G4H / 256, 256, 0, stream>>>(b_ih1, b_hh1, bias1, G4H);
  k_gv<<<(B_ * GE_) / 256, 256, 0, stream>>>(gemb, gids, gv);
  k_state<<<(B_ * H_) / 256, 256, 0, stream>>>(h0in, c0in, h0A, h1A, c0w, c1w);
  k_wt<<<(K0PAD * G4H) / 256, 256, 0, stream>>>(W_ih0, wt_ih0, IN0_, K0PAD * G4H);
  k_wt<<<(H_ * G4H) / 256, 256, 0, stream>>>(W_hh0, wt_hh0, H_, H_ * G4H);
  k_wt<<<(H_ * G4H) / 256, 256, 0, stream>>>(W_ih1, wt_ih1, H_, H_ * G4H);
  k_wt<<<(H_ * G4H) / 256, 256, 0, stream>>>(W_hh1, wt_hh1, H_, H_ * G4H);

  // ---- main recurrent kernel ----
  Params p;
  p.wt_ih0 = wt_ih0; p.wt_hh0 = wt_hh0; p.wt_ih1 = wt_ih1; p.wt_hh1 = wt_hh1;
  p.h0buf[0] = h0A; p.h0buf[1] = h0B;
  p.h1buf[0] = h1A; p.h1buf[1] = h1B;
  p.c0 = c0w; p.c1 = c1w;
  p.gv = gv; p.bias0 = bias0; p.bias1 = bias1;
  p.predpart = predpart; p.bar = bar;
  p.dec_known = dec_known; p.target_y = target_y; p.last_enc = last_enc;
  p.W_proj = W_proj; p.b_proj = b_proj; p.tf = tf;
  p.out = (float*)d_out;

  (void)hipFuncSetAttribute((const void*)lstm_main,
                            hipFuncAttributeMaxDynamicSharedMemorySize,
                            SMEM_BYTES);
  lstm_main<<<NBT * CG, NTHREADS, SMEM_BYTES, stream>>>(p);
}
